// QManoLayer_34084860461629
// MI455X (gfx1250) — matrix operand order repaired and checked by reading, not compiled, not run on hardware
//
#include <hip/hip_runtime.h>
#include <hip/hip_bf16.h>

// ---------------------------------------------------------------------------
// MANO forward pass for MI455X (gfx1250, wave32, WMMA)
//   B=8192 batches, V=778 vertices, J=16 joints, 45 PCA pose comps.
//
// Skinning GEMM on v_wmma_f32_16x16x32_f16 with TRANSPOSED roles:
//   D[mn, v] = R2^T[mn, j] x W^T[j, v]
// A = per-batch transform block (pre-swizzled f16 A-fragments, 1 KB/batch,
//     coalesced 32B/lane loads, DOUBLE-BUFFERED one full iteration ahead),
// B = constant weight tile (built once per wave).
// D layout puts all 8 skinT components of one vertex in ONE lane, so the
// rest_h application is an in-lane dot product (no cross-lane traffic).
// All per-batch operands walk raw pointers (no per-iter index math).
// ---------------------------------------------------------------------------

#define V_  778
#define J_  16
#define B_  8192
#define NC  45
#define NTILE 49              // ceil(778/16)
#define BATCH_PER_WAVE 32

typedef _Float16 half_t;
typedef __attribute__((ext_vector_type(16))) _Float16 v16h;
typedef __attribute__((ext_vector_type(8)))  float    v8f;

// ws layout: [0, B*1024) swizzled f16 A-fragments (512 halfs per batch),
// then JRSD (480 f32), JRT (48 f32).  The JRSD/JRT tail (2112 B) doubles as
// the in-bounds overread target for the double-buffered lookahead loads.
#define WS_AFRAG_BYTES ((size_t)B_ * 1024)

// ---------------------------------------------------------------------------
// Kernel 0: fold J_regressor into shapedirs/template (one tiny block).
// JRSD[j,d,s] = sum_v JR[j,v]*SD[v,d,s]; JRT[j,d] = JR@tmpl.
// ---------------------------------------------------------------------------
__global__ void precompute_jr(const float* __restrict__ JR,
                              const float* __restrict__ SD,
                              const float* __restrict__ tmpl,
                              float* __restrict__ JRSD,
                              float* __restrict__ JRT) {
  int t = threadIdx.x;
  if (t < 480) {
    int s = t % 10, jd = t / 10;
    int j = jd / 3, d = jd % 3;
    float acc = 0.f;
    for (int v = 0; v < V_; ++v)
      acc += JR[j * V_ + v] * SD[(v * 3 + d) * 10 + s];
    JRSD[jd * 10 + s] = acc;
  } else if (t < 528) {
    int jd = t - 480;
    int j = jd / 3, d = jd % 3;
    float acc = 0.f;
    for (int v = 0; v < V_; ++v)
      acc += JR[j * V_ + v] * tmpl[v * 3 + d];
    JRT[jd] = acc;
  }
}

// 3x4 affine (implicit [0,0,0,1] row): G = G * Rl
__device__ __forceinline__ void amul(float* G, const float* Rl) {
  float C[12];
#pragma unroll
  for (int m = 0; m < 3; ++m) {
#pragma unroll
    for (int n = 0; n < 3; ++n)
      C[m * 4 + n] = G[m * 4 + 0] * Rl[0 * 4 + n] +
                     G[m * 4 + 1] * Rl[1 * 4 + n] +
                     G[m * 4 + 2] * Rl[2 * 4 + n];
    C[m * 4 + 3] = G[m * 4 + 0] * Rl[3] + G[m * 4 + 1] * Rl[7] +
                   G[m * 4 + 2] * Rl[11] + G[m * 4 + 3];
  }
#pragma unroll
  for (int i = 0; i < 12; ++i) G[i] = C[i];
}

// ---------------------------------------------------------------------------
// Kernel 1: lane-per-joint. 256 threads = 16 batches x 16 joints.
// PCA pose -> Rodrigues -> kinematic chain (REORDER is identity on joint id)
// -> results2. Emits jtr[:, :16] and the swizzled f16 A-fragment per batch.
//
// A-matrix (16x32 f16): M = mn column index (m*4+n), K = joint j (16..31 pad).
// ISA A layout: lane L (lhi=L/16), element e: M=L%16, K = e + 8*lhi  (e<8)
//                                             K = (e-8) + 8*lhi + 16 (e>=8).
// Data (K<16) lives at: j<8 -> lanes 0-15 e=j; j>=8 -> lanes 16-31 e=j-8.
// ---------------------------------------------------------------------------
__global__ void pose_chain(const float* __restrict__ pose,
                           const float* __restrict__ betas,
                           const float* __restrict__ trans,
                           const float* __restrict__ SC,
                           const float* __restrict__ hmean,
                           const float* __restrict__ JRSD,
                           const float* __restrict__ JRT,
                           half_t* __restrict__ wsA,
                           float* __restrict__ jtr) {
  __shared__ float sJ[16][J_][3];
  __shared__ float sRel[16][J_][12];

  int tid = threadIdx.x;
  int bl = tid >> 4, j = tid & 15;
  int b = blockIdx.x * 16 + bl;

  // full_pose (axis-angle) for this joint
  float fp0, fp1, fp2;
  if (j == 0) {
    fp0 = pose[b * 48 + 0]; fp1 = pose[b * 48 + 1]; fp2 = pose[b * 48 + 2];
  } else {
    int c0 = 3 * (j - 1);
    float a0 = hmean[c0], a1 = hmean[c0 + 1], a2 = hmean[c0 + 2];
    for (int k = 0; k < NC; ++k) {
      float p = pose[b * 48 + 3 + k];
      a0 += p * SC[k * 45 + c0];
      a1 += p * SC[k * 45 + c0 + 1];
      a2 += p * SC[k * 45 + c0 + 2];
    }
    fp0 = a0; fp1 = a1; fp2 = a2;
  }

  // Rodrigues (matches reference: angle = ||aa + 1e-8||, axis = aa/angle)
  float ex = fp0 + 1e-8f, ey = fp1 + 1e-8f, ez = fp2 + 1e-8f;
  float ang = sqrtf(ex * ex + ey * ey + ez * ez);
  float inv = 1.f / ang;
  float ha = 0.5f * ang;
  float cw = cosf(ha), sn = sinf(ha);
  float x = fp0 * inv * sn, y = fp1 * inv * sn, z = fp2 * inv * sn;
  float R[9];
  R[0] = 1.f - 2.f * (y * y + z * z); R[1] = 2.f * (x * y - cw * z); R[2] = 2.f * (x * z + cw * y);
  R[3] = 2.f * (x * y + cw * z); R[4] = 1.f - 2.f * (x * x + z * z); R[5] = 2.f * (y * z - cw * x);
  R[6] = 2.f * (x * z - cw * y); R[7] = 2.f * (y * z + cw * x); R[8] = 1.f - 2.f * (x * x + y * y);

  // joint rest position via folded regressor
  float jv[3];
#pragma unroll
  for (int d = 0; d < 3; ++d) {
    float acc = JRT[j * 3 + d];
#pragma unroll
    for (int s = 0; s < 10; ++s) acc += JRSD[(j * 3 + d) * 10 + s] * betas[b * 10 + s];
    jv[d] = acc;
    sJ[bl][j][d] = acc;
  }
  __syncthreads();

  // relative transform for this joint
  float t3[3];
  if (j == 0) {
    t3[0] = jv[0]; t3[1] = jv[1]; t3[2] = jv[2];
  } else {
    int p = (((j - 1) % 3) == 0) ? 0 : (j - 1);
    t3[0] = jv[0] - sJ[bl][p][0];
    t3[1] = jv[1] - sJ[bl][p][1];
    t3[2] = jv[2] - sJ[bl][p][2];
  }
  float* rl = sRel[bl][j];
  rl[0] = R[0]; rl[1] = R[1]; rl[2]  = R[2]; rl[3]  = t3[0];
  rl[4] = R[3]; rl[5] = R[4]; rl[6]  = R[5]; rl[7]  = t3[1];
  rl[8] = R[6]; rl[9] = R[7]; rl[10] = R[8]; rl[11] = t3[2];
  __syncthreads();

  // global transform: G = rel[0] * rel[base] * ... * rel[j]
  float G[12];
#pragma unroll
  for (int i = 0; i < 12; ++i) G[i] = sRel[bl][0][i];
  if (j > 0) {
    int base = ((j - 1) / 3) * 3 + 1;
    for (int q = base; q <= j; ++q) amul(G, sRel[bl][q]);
  }

  // jtr[:, :16] = G.t + trans
#pragma unroll
  for (int d = 0; d < 3; ++d)
    jtr[((size_t)b * 21 + j) * 3 + d] = G[d * 4 + 3] + trans[b * 3 + d];

  // results2 translation column: t2 = G.t - G.R * jv
  float t2[3];
#pragma unroll
  for (int m = 0; m < 3; ++m)
    t2[m] = G[m * 4 + 3] -
            (G[m * 4 + 0] * jv[0] + G[m * 4 + 1] * jv[1] + G[m * 4 + 2] * jv[2]);

  // emit swizzled f16 A fragment: A[mn, j] = results2[j, m, n]
  half_t* wa = wsA + (size_t)b * 512;
#pragma unroll
  for (int c = 0; c < 16; ++c) {        // c = mn = m*4 + n
    int m = c >> 2, nn = c & 3;
    float val;
    if (m < 3) val = (nn < 3) ? G[m * 4 + nn] : t2[m];
    else       val = (nn == 3) ? 1.f : 0.f;
    half_t hv = (half_t)val;
    if (j < 8) {
      wa[c * 16 + j] = hv;                      // lhi=0 data, e=j
      wa[(16 + c) * 16 + 8 + j] = (half_t)0.f;  // lhi=1 pad,  e=8+j
    } else {
      wa[(16 + c) * 16 + (j - 8)] = hv;         // lhi=1 data, e=j-8
      wa[c * 16 + j] = (half_t)0.f;             // lhi=0 pad,  e=j
    }
  }
}

// ---------------------------------------------------------------------------
// Kernel 2: WMMA skinning. One wave = one 16-vertex tile x 32 batches.
//   D(16mn x 16v) = A(16mn x 32j per-batch transforms) x B(32j x 16v weights)
// Lane L owns vertex v0 + L%16: lanes 0-15 hold skinT rows mn=0..7 (m=0,1),
// lanes 16-31 hold mn=8..15 (m=2,[3]). rest_h applied as an in-lane dot.
// Fragment loads double-buffered: each load has a full body to complete.
// ---------------------------------------------------------------------------
__global__ void __launch_bounds__(256)
skin_verts(const float* __restrict__ W, const float* __restrict__ betas,
           const float* __restrict__ trans, const float* __restrict__ tmpl,
           const float* __restrict__ SD, const half_t* __restrict__ wsA,
           float* __restrict__ verts) {
  int lane = threadIdx.x & 31;
  int gw = blockIdx.x * (blockDim.x >> 5) + (threadIdx.x >> 5);
  int vtile = gw % NTILE;
  int b0 = (gw / NTILE) * BATCH_PER_WAVE;
  int v0 = vtile * 16;
  int lv = lane & 15;
  int lhi = lane >> 4;
  int v = v0 + lv;
  int vc = (v < V_) ? v : (V_ - 1);
  bool vok = (v < V_);
  int trOff = lhi ? 2 : 0;

  // B fragment (constant): B[k=j, n=v] = W[v, j]; K=16..31 is zero padding.
  // B layout: lane L, element e -> B[k = e + 8*lhi (e<8), k = e - 8 + 8*lhi + 16 (e>=8), n = L%16].
  v16h bw;
#pragma unroll
  for (int e = 0; e < 16; ++e) {
    float w = (e < 8 && vok) ? W[v * 16 + 8 * lhi + e] : 0.f;
    bw[e] = (half_t)w;
  }

  // hoist per-vertex shape data (batch-invariant): 33 VGPRs
  float tm[3], SDr[3][10];
#pragma unroll
  for (int n = 0; n < 3; ++n) {
    tm[n] = tmpl[vc * 3 + n];
#pragma unroll
    for (int s = 0; s < 10; ++s) SDr[n][s] = SD[(vc * 3 + n) * 10 + s];
  }

  // raw pointer walks: no per-iteration index math
  const half_t* ap = wsA + (size_t)b0 * 512 + lane * 16;
  const float*  bp = betas + b0 * 10;
  const float*  tp = trans + b0 * 3;
  float*        vp = verts + ((size_t)b0 * V_ + v) * 3;

  auto body = [&](v16h afc) {
    // per-batch scalars issued early; they hide under the WMMA
    float bet[10];
#pragma unroll
    for (int s = 0; s < 10; ++s) bet[s] = bp[s];
    float trA = tp[trOff];
    float trB = tp[1];

    v8f c = {};
    c = __builtin_amdgcn_wmma_f32_16x16x32_f16(
        /*neg_a=*/false, afc, /*neg_b=*/false, bw,
        /*c_mod=*/(short)0, c, /*reuse_a=*/false, /*reuse_b=*/false);

    // rest_h for this lane's single vertex (registers only)
    float rh[4];
#pragma unroll
    for (int n = 0; n < 3; ++n) {
      float acc = tm[n];
#pragma unroll
      for (int s = 0; s < 10; ++s) acc += SDr[n][s] * bet[s];
      rh[n] = acc;
    }
    rh[3] = 1.f;

    // rows r=0..3 -> m = 2*lhi, rows 4..7 -> m = 2*lhi+1
    float o0 = c[0] * rh[0] + c[1] * rh[1] + c[2] * rh[2] + c[3] * rh[3];
    float o1 = c[4] * rh[0] + c[5] * rh[1] + c[6] * rh[2] + c[7] * rh[3];

    if (vok) {
      vp[trOff] = o0 + trA;                 // coord 0 (lanes 0-15) or 2
      if (lhi == 0) vp[1] = o1 + trB;       // coord 1 (lanes 0-15 only)
    }
    bp += 10; tp += 3; vp += 3 * V_;
  };

  // double-buffered pipeline; final lookahead loads land in the in-bounds
  // JRSD/JRT tail of ws and are never consumed.
  v16h afA = *(const v16h*)ap;
  for (int bi = 0; bi < BATCH_PER_WAVE; bi += 2) {
    ap += 512;
    v16h afB = *(const v16h*)ap;
    __builtin_prefetch(ap + 512, 0, 3);
    body(afA);
    ap += 512;
    afA = *(const v16h*)ap;
    __builtin_prefetch(ap + 512, 0, 3);
    body(afB);
  }
}

// ---------------------------------------------------------------------------
// Kernel 3: fingertip joints — jtr[:,16+k] = verts[:, tips[k]] (verts already
// carry +trans, matching the reference ordering of additions).
// ---------------------------------------------------------------------------
__global__ void tips_kernel(const float* __restrict__ verts, float* __restrict__ jtr) {
  int t = blockIdx.x * blockDim.x + threadIdx.x;
  if (t >= B_ * 5) return;
  int b = t / 5, k = t % 5;
  int v;
  switch (k) {
    case 0: v = 745; break;
    case 1: v = 317; break;
    case 2: v = 444; break;
    case 3: v = 556; break;
    default: v = 673; break;
  }
#pragma unroll
  for (int d = 0; d < 3; ++d)
    jtr[((size_t)b * 21 + 16 + k) * 3 + d] = verts[((size_t)b * V_ + v) * 3 + d];
}

extern "C" void kernel_launch(void* const* d_in, const int* in_sizes, int n_in,
                              void* d_out, int out_size, void* d_ws, size_t ws_size,
                              hipStream_t stream) {
  (void)in_sizes; (void)n_in; (void)out_size; (void)ws_size;
  const float* pose  = (const float*)d_in[0];
  const float* betas = (const float*)d_in[1];
  const float* trans = (const float*)d_in[2];
  const float* SC    = (const float*)d_in[3];
  const float* hmean = (const float*)d_in[4];
  const float* tmpl  = (const float*)d_in[5];
  const float* SD    = (const float*)d_in[6];
  const float* JR    = (const float*)d_in[7];
  const float* W     = (const float*)d_in[8];

  float* verts = (float*)d_out;
  float* jtr   = verts + (size_t)B_ * V_ * 3;

  half_t* wsA = (half_t*)d_ws;
  float* JRSD = (float*)((char*)d_ws + WS_AFRAG_BYTES);
  float* JRT  = JRSD + 480;

  precompute_jr<<<1, 544, 0, stream>>>(JR, SD, tmpl, JRSD, JRT);
  pose_chain<<<B_ / 16, 256, 0, stream>>>(pose, betas, trans, SC, hmean,
                                          JRSD, JRT, wsA, jtr);
  int nblocks = (NTILE * (B_ / BATCH_PER_WAVE)) / 8;   // 1568 blocks, 8 waves
  skin_verts<<<nblocks, 256, 0, stream>>>(W, betas, trans, tmpl, SD, wsA, verts);
  tips_kernel<<<(B_ * 5 + 255) / 256, 256, 0, stream>>>(verts, jtr);
}
